// ImportancePooling_5368709120465
// MI455X (gfx1250) — matrix operand order repaired and checked by reading, not compiled, not run on hardware
//
#include <hip/hip_runtime.h>

#define NUM_TARGET 20000
#define KNBR 32
#define DIM 256
#define HID 128
#define TGT_PER_WAVE 2
#define WAVES_PER_BLK 4

typedef __attribute__((ext_vector_type(16))) _Float16 v16h;
typedef __attribute__((ext_vector_type(8)))  float    v8f;

// ---------------------------------------------------------------------------
// Kernel A: one-time f32 -> f16 conversion of W1 into workspace.
// ---------------------------------------------------------------------------
__global__ __launch_bounds__(256)
void convert_w1_kernel(const float* __restrict__ W1, _Float16* __restrict__ W1h) {
    const int i = blockIdx.x * 256 + threadIdx.x;   // DIM*HID = 32768 elements
    W1h[i] = (_Float16)W1[i];
}

// A-fragment (16x32 f16, M x K) per CDNA5 ISA layout:
//   lanes 0-15: M = lane,    K in {kbase+0..7, kbase+16..23}
//   lanes16-31: M = lane-16, K in {kbase+8..15, kbase+24..31}
__device__ __forceinline__ v16h load_a_frag(const float* __restrict__ row, int kbase, int lane) {
    const int off = kbase + ((lane & 16) ? 8 : 0);
    v16h a;
#pragma unroll
    for (int j = 0; j < 8; ++j) a[j]     = (_Float16)row[off + j];
#pragma unroll
    for (int j = 0; j < 8; ++j) a[8 + j] = (_Float16)row[off + 16 + j];
    return a;
}

// ---------------------------------------------------------------------------
// Kernel B: blended neighbor weights. One wave handles TWO targets so every
// B-fragment (W1h) load feeds 4 WMMAs. 128 WMMA / target.
// B-fragment (32x16 f16, K x N): N = lane % 16; 16 f16 per lane = K, same order as A;
// with W1h row-major [DIM,HID] that is 16 scalar loads, HID apart.
// ---------------------------------------------------------------------------
__global__ __launch_bounds__(WAVES_PER_BLK * 32)
void pinsage_scores_kernel(const float*    __restrict__ x,
                           const int*      __restrict__ nbr_idx,
                           const float*    __restrict__ imp_w,
                           const _Float16* __restrict__ W1h,
                           const float*    __restrict__ b1,
                           const float*    __restrict__ W2,
                           const float*    __restrict__ b2,
                           float*          __restrict__ wout) {
    const int lane = threadIdx.x & 31;
    const int wid  = threadIdx.x >> 5;
    const int nl   = lane & 15;
    const int n0   = (blockIdx.x * WAVES_PER_BLK + wid) * TGT_PER_WAVE;  // target pair

    __shared__ float sc[WAVES_PER_BLK][TGT_PER_WAVE][KNBR];

    // Row pointers: 4 M-tiles = {t0 rows 0-15, t0 rows 16-31, t1 rows 0-15, t1 rows 16-31}
    const int* nb0 = nbr_idx + (size_t)n0 * KNBR;
    const float* rowp[4];
    rowp[0] = x + (size_t)nb0[nl]            * DIM;
    rowp[1] = x + (size_t)nb0[16 + nl]       * DIM;
    rowp[2] = x + (size_t)nb0[KNBR + nl]     * DIM;
    rowp[3] = x + (size_t)nb0[KNBR + 16 + nl]* DIM;

    // H accumulators: 4 M-tiles x 8 N-tiles  (256 VGPRs)
    v8f acc[4][8];
#pragma unroll
    for (int mi = 0; mi < 4; ++mi)
#pragma unroll
        for (int ni = 0; ni < 8; ++ni) { v8f z = {}; acc[mi][ni] = z; }

    // GEMM: [64,256] x [256,128], K in 8 steps of 32 -> 256 WMMAs per wave
#pragma unroll
    for (int ks = 0; ks < 8; ++ks) {
        v16h a[4];
#pragma unroll
        for (int mi = 0; mi < 4; ++mi) a[mi] = load_a_frag(rowp[mi], ks * 32, lane);
#pragma unroll
        for (int ni = 0; ni < 8; ++ni) {
            const int kb = ks * 32 + ((lane & 16) ? 8 : 0);
            v16h b;
#pragma unroll
            for (int j = 0; j < 8; ++j) b[j]     = W1h[(size_t)(kb + j) * HID + ni * 16 + nl];
#pragma unroll
            for (int j = 0; j < 8; ++j) b[8 + j] = W1h[(size_t)(kb + 16 + j) * HID + ni * 16 + nl];
#pragma unroll
            for (int mi = 0; mi < 4; ++mi) {
                acc[mi][ni] = __builtin_amdgcn_wmma_f32_16x16x32_f16(
                    false, a[mi], false, b, (short)0, acc[mi][ni], false, false);
            }
        }
    }

    // scores = relu(H + b1) @ W2 + b2   (N=1 contraction on VALU)
    // C-layout: VGPR r, lanes 0-15: (M=r, N=lane); lanes 16-31: (M=8+r, N=lane-16)
    float b1v[8], w2v[8];
#pragma unroll
    for (int ni = 0; ni < 8; ++ni) {
        b1v[ni] = b1[ni * 16 + nl];
        w2v[ni] = W2[ni * 16 + nl];
    }
    const float bias2 = b2[0];

#pragma unroll
    for (int t = 0; t < TGT_PER_WAVE; ++t) {
        float p[16];
#pragma unroll
        for (int mi = 0; mi < 2; ++mi)
#pragma unroll
            for (int r = 0; r < 8; ++r) {
                float s = 0.f;
#pragma unroll
                for (int ni = 0; ni < 8; ++ni) {
                    float h = acc[t * 2 + mi][ni][r] + b1v[ni];
                    h = fmaxf(h, 0.f);
                    s += h * w2v[ni];
                }
                p[mi * 8 + r] = s;
            }
        // Reduce the N dimension (spread over the 16 lanes of each half-wave)
#pragma unroll
        for (int off = 1; off < 16; off <<= 1)
#pragma unroll
            for (int j = 0; j < 16; ++j) p[j] += __shfl_xor(p[j], off, 32);

        if (lane == 0) {
#pragma unroll
            for (int mi = 0; mi < 2; ++mi)
#pragma unroll
                for (int r = 0; r < 8; ++r) sc[wid][t][mi * 16 + r] = p[mi * 8 + r] + bias2;
        }
        if (lane == 16) {
#pragma unroll
            for (int mi = 0; mi < 2; ++mi)
#pragma unroll
                for (int r = 0; r < 8; ++r) sc[wid][t][mi * 16 + 8 + r] = p[mi * 8 + r] + bias2;
        }
    }
    __syncthreads();

    // Per-lane k = lane: softmax(learned) blended with softmax(imp_w)
#pragma unroll
    for (int t = 0; t < TGT_PER_WAVE; ++t) {
        const int n = n0 + t;
        float s  = sc[wid][t][lane];
        float mx = s;
#pragma unroll
        for (int off = 16; off >= 1; off >>= 1) mx = fmaxf(mx, __shfl_xor(mx, off, 32));
        float e  = __expf(s - mx);
        float es = e;
#pragma unroll
        for (int off = 16; off >= 1; off >>= 1) es += __shfl_xor(es, off, 32);
        const float learned = e / es;

        float iw = imp_w[(size_t)n * KNBR + lane];
        float m2 = iw;
#pragma unroll
        for (int off = 16; off >= 1; off >>= 1) m2 = fmaxf(m2, __shfl_xor(m2, off, 32));
        float e2  = __expf(iw - m2);
        float es2 = e2;
#pragma unroll
        for (int off = 16; off >= 1; off >>= 1) es2 += __shfl_xor(es2, off, 32);
        const float pre = e2 / es2;

        wout[(size_t)n * KNBR + lane] = 0.5f * learned + 0.5f * pre;
    }
}

// ---------------------------------------------------------------------------
// Kernel C: full-f32 weighted gather-sum: out[n,:] = sum_k w[n,k]*x[nbr[n,k],:]
// x is L2-resident (102 MB < 192 MB L2); float4-coalesced along D.
// ---------------------------------------------------------------------------
__global__ __launch_bounds__(64)
void pinsage_pool_kernel(const float* __restrict__ x,
                         const int*   __restrict__ nbr_idx,
                         const float* __restrict__ w,
                         float*       __restrict__ out) {
    const int n = blockIdx.x;
    const int t = threadIdx.x;          // 0..63, 4 floats each
    __shared__ int   sidx[KNBR];
    __shared__ float sw[KNBR];
    if (t < KNBR) {
        sidx[t] = nbr_idx[(size_t)n * KNBR + t];
        sw[t]   = w[(size_t)n * KNBR + t];
    }
    __syncthreads();

    const int d = t * 4;
    float4 acc = make_float4(0.f, 0.f, 0.f, 0.f);
#pragma unroll
    for (int k = 0; k < KNBR; ++k) {
        const float4 v  = *(const float4*)(x + (size_t)sidx[k] * DIM + d);
        const float  wk = sw[k];
        acc.x += wk * v.x; acc.y += wk * v.y; acc.z += wk * v.z; acc.w += wk * v.w;
    }
    *(float4*)(out + (size_t)n * DIM + d) = acc;
}

extern "C" void kernel_launch(void* const* d_in, const int* in_sizes, int n_in,
                              void* d_out, int out_size, void* d_ws, size_t ws_size,
                              hipStream_t stream) {
    const float* x    = (const float*)d_in[0];
    const int*   nbr  = (const int*)  d_in[1];
    const float* imp  = (const float*)d_in[2];
    const float* W1   = (const float*)d_in[3];
    const float* b1   = (const float*)d_in[4];
    const float* W2   = (const float*)d_in[5];
    const float* b2   = (const float*)d_in[6];

    // Workspace layout: [0, 64KB)  W1 in f16
    //                   [64KB, +2.56MB) blended weights
    _Float16* W1h  = (_Float16*)d_ws;
    float*    wbuf = (float*)((char*)d_ws + (size_t)DIM * HID * sizeof(_Float16));

    convert_w1_kernel<<<(DIM * HID) / 256, 256, 0, stream>>>(W1, W1h);

    const int waves  = NUM_TARGET / TGT_PER_WAVE;          // 10000
    const int blocks = waves / WAVES_PER_BLK;              // 2500
    pinsage_scores_kernel<<<blocks, WAVES_PER_BLK * 32, 0, stream>>>(
        x, nbr, imp, W1h, b1, W2, b2, wbuf);

    pinsage_pool_kernel<<<NUM_TARGET, 64, 0, stream>>>(x, nbr, wbuf, (float*)d_out);
}
